// WindowCrossAttention_25159918420368
// MI455X (gfx1250) — matrix operand order repaired and checked by reading, not compiled, not run on hardware
//
#include <hip/hip_runtime.h>

// WindowCrossAttention for MI455X (gfx1250, wave32, WMMA f16 16x16x32).
// One 256-thread workgroup (8 waves) per batch element; wave h owns head h.
// f16 inputs / f32 accumulate; softmax in f32; final output f32.

#define NTOK 49
#define NP   64
#define CDIM 256
#define HEADS 8
#define HD   32

typedef __attribute__((ext_vector_type(16))) _Float16 v16h;
typedef __attribute__((ext_vector_type(8)))  _Float16 v8h;
typedef __attribute__((ext_vector_type(4)))  _Float16 v4h;
typedef __attribute__((ext_vector_type(8)))  float    v8f;
typedef __attribute__((ext_vector_type(4)))  float    v4f;

static __device__ __forceinline__ v8f wmma16(v16h a, v16h b, v8f c) {
  // (neg_a, A, neg_b, B, c_mod, C, reuse_a, reuse_b)
  return __builtin_amdgcn_wmma_f32_16x16x32_f16(false, a, false, b, (short)0, c,
                                                false, false);
}

// A-matrix fragment (16x32 f16), row-major source, row stride ld (halves).
// Lane layout (ISA 7.12.2): lanes 0-15 hold K=[0..7]+[16..23], lanes 16-31
// hold K=[8..15]+[24..31] for row M = lane%16.
static __device__ __forceinline__ v16h load_a16(const _Float16* base, int ld, int lane) {
  const int m = lane & 15, ks = (lane >> 4) << 3;
  const _Float16* p = base + m * ld + ks;
  v8h lo = *(const v8h*)p;          // K = ks..ks+7
  v8h hi = *(const v8h*)(p + 16);   // K = 16+ks..
  v16h r;
#pragma unroll
  for (int i = 0; i < 8; ++i) { r[i] = lo[i]; r[8 + i] = hi[i]; }
  return r;
}

// B-matrix fragment (32x16 f16) from "n-major" source: src[n*ld + k].
// Lane layout: lanes 0-15 hold K=[0..7]+[16..23], lanes 16-31 hold
// K=[8..15]+[24..31] of column n = lane%16 (same K order as load_a16).
static __device__ __forceinline__ v16h load_b16(const _Float16* base, int ld, int lane) {
  const int n = lane & 15, ks = (lane >> 4) << 3;
  const _Float16* p = base + n * ld + ks;
  v8h lo = *(const v8h*)p;
  v8h hi = *(const v8h*)(p + 16);
  v16h r;
#pragma unroll
  for (int i = 0; i < 8; ++i) { r[i] = lo[i]; r[8 + i] = hi[i]; }
  return r;
}

// Y(64x32) = X(64x256) @ W[32 rows x 256]^T + bvec, scaled; store f16 to LDS,
// row-major (ld=32) or transposed (dst[col*64 + row]).
static __device__ __forceinline__ void gemm_px(
    const _Float16* X, const _Float16* W, const float* bvec, float scale,
    _Float16* dst, bool transposed, int lane) {
  const int ln = lane & 15, hl = lane >> 4;
  for (int mt = 0; mt < 4; ++mt) {
    v8f acc0 = {}, acc1 = {};
#pragma unroll
    for (int kt = 0; kt < 8; ++kt) {
      v16h a  = load_a16(X + mt * 16 * CDIM + kt * 32, CDIM, lane);
      v16h b0 = load_b16(W + kt * 32, CDIM, lane);
      v16h b1 = load_b16(W + 16 * CDIM + kt * 32, CDIM, lane);
      acc0 = wmma16(a, b0, acc0);
      acc1 = wmma16(a, b1, acc1);
    }
    const float bb0 = bvec[ln], bb1 = bvec[16 + ln];
#pragma unroll
    for (int r = 0; r < 8; ++r) {
      const int row = mt * 16 + r + hl * 8;
      const float y0 = (acc0[r] + bb0) * scale;
      const float y1 = (acc1[r] + bb1) * scale;
      if (!transposed) {
        dst[row * HD + ln]      = (_Float16)y0;
        dst[row * HD + 16 + ln] = (_Float16)y1;
      } else {
        dst[ln * NP + row]        = (_Float16)y0;
        dst[(16 + ln) * NP + row] = (_Float16)y1;
      }
    }
  }
}

// Prep: fp32->f16 weights in workspace; pre-gathered padded bias table
// bias[h][64][64], with -1e30 in key columns >= 49 (softmax mask fused in).
extern "C" __global__ void winxattn_prep(
    const float* __restrict__ qkv_w, const float* __restrict__ proj_w,
    const float* __restrict__ rpb, const int* __restrict__ rel,
    _Float16* __restrict__ w16, _Float16* __restrict__ pw16,
    float* __restrict__ biasp) {
  const int stride = gridDim.x * blockDim.x;
  const int tid = blockIdx.x * blockDim.x + threadIdx.x;
  for (int i = tid; i < 3 * CDIM * CDIM; i += stride) w16[i] = (_Float16)qkv_w[i];
  for (int i = tid; i < CDIM * CDIM; i += stride)     pw16[i] = (_Float16)proj_w[i];
  for (int i = tid; i < HEADS * NP * NP; i += stride) {
    const int h = i >> 12, n = (i >> 6) & 63, m = i & 63;
    float v;
    if (m >= NTOK)      v = -1e30f;           // mask padded key columns
    else if (n >= NTOK) v = 0.0f;             // padded query rows: don't care
    else                v = rpb[rel[n * NTOK + m] * HEADS + h];
    biasp[i] = v;
  }
}

extern "C" __global__ void __launch_bounds__(256, 1)
winxattn_main(const float* __restrict__ x1, const float* __restrict__ x2,
              const float* __restrict__ qkv_b, const float* __restrict__ proj_b,
              const _Float16* __restrict__ w16, const _Float16* __restrict__ pw16,
              const float* __restrict__ biasp, float* __restrict__ out) {
  extern __shared__ _Float16 smem[];
  const int tid  = threadIdx.x;
  const int lane = tid & 31;
  const int h    = tid >> 5;          // wave id == head id
  const int ln   = lane & 15;
  const int hl   = lane >> 4;

  // LDS carve (262144 B total):
  _Float16* Xa  = smem;                         // 64x256 f16  (32 KB)
  _Float16* Xb  = Xa + NP * CDIM;               // 64x256 f16  (32 KB)
  _Float16* hb  = Xb + NP * CDIM + h * 10240;   // per-head 20 KB
  _Float16* Qh  = hb;                           // 64x32
  _Float16* Kh  = hb + 2048;                    // 64x32
  _Float16* VhT = hb + 4096;                    // 32x64 (transposed V)
  _Float16* Ph  = hb + 6144;                    // 64x64 probs
  _Float16* Osh = Xb + NP * CDIM + HEADS * 10240; // 64x256 f16 (32 KB)

  const size_t xoff = (size_t)blockIdx.x * (NTOK * CDIM);
  const float* px1 = x1 + xoff;
  const float* px2 = x2 + xoff;

  // ---- Stage x1/x2 -> f16 LDS, zero-fill padded rows [49,64) ----
  for (int i = tid; i < NP * (CDIM / 4); i += 256) {
    const int row = i >> 6;
    const int c4  = (i & 63) << 2;
    v4f a = {0.f, 0.f, 0.f, 0.f}, b = {0.f, 0.f, 0.f, 0.f};
    if (row < NTOK) {
      a = *(const v4f*)(px1 + row * CDIM + c4);
      b = *(const v4f*)(px2 + row * CDIM + c4);
    }
    v4h ah, bh;
#pragma unroll
    for (int j = 0; j < 4; ++j) { ah[j] = (_Float16)a[j]; bh[j] = (_Float16)b[j]; }
    *(v4h*)(Xa + row * CDIM + c4) = ah;
    *(v4h*)(Xb + row * CDIM + c4) = bh;
  }
  __syncthreads();

  // ---- Q/K/V projections for this head (scale folded into Q) ----
  const float kscale = 0.17677669529663687f;  // 1/sqrt(32)
  gemm_px(Xa, w16 + (size_t)(h * HD) * CDIM,              qkv_b + h * HD,              kscale, Qh,  false, lane);
  gemm_px(Xb, w16 + (size_t)(CDIM + h * HD) * CDIM,       qkv_b + CDIM + h * HD,       1.0f,   Kh,  false, lane);
  gemm_px(Xb, w16 + (size_t)(2 * CDIM + h * HD) * CDIM,   qkv_b + 2 * CDIM + h * HD,   1.0f,   VhT, true,  lane);

  // ---- S = Q K^T + bias ; softmax ; P -> LDS (f16) ----
  const float* bp = biasp + h * NP * NP;
  for (int mt = 0; mt < 4; ++mt) {
    v16h aq = load_a16(Qh + mt * 16 * HD, HD, lane);   // K-dim = 32, one step
    v8f s[4];
#pragma unroll
    for (int nt = 0; nt < 4; ++nt) {
      v16h bk = load_b16(Kh + nt * 16 * HD, HD, lane); // column n = key row
      v8f z = {};
      s[nt] = wmma16(aq, bk, z);
      const int col = nt * 16 + ln;
#pragma unroll
      for (int r = 0; r < 8; ++r)
        s[nt][r] += bp[(mt * 16 + r + hl * 8) * NP + col];
    }
    float inv[8];
#pragma unroll
    for (int r = 0; r < 8; ++r) {
      float m0 = fmaxf(fmaxf(s[0][r], s[1][r]), fmaxf(s[2][r], s[3][r]));
#pragma unroll
      for (int k = 1; k < 16; k <<= 1) m0 = fmaxf(m0, __shfl_xor(m0, k, 32));
      float t = 0.f;
#pragma unroll
      for (int nt = 0; nt < 4; ++nt) { s[nt][r] = __expf(s[nt][r] - m0); t += s[nt][r]; }
#pragma unroll
      for (int k = 1; k < 16; k <<= 1) t += __shfl_xor(t, k, 32);
      inv[r] = 1.0f / t;
    }
#pragma unroll
    for (int nt = 0; nt < 4; ++nt)
#pragma unroll
      for (int r = 0; r < 8; ++r)
        Ph[(mt * 16 + r + hl * 8) * NP + nt * 16 + ln] =
            (_Float16)(s[nt][r] * inv[r]);
  }

  // ---- O = P V ; write f16 into shared 64x256 O buffer at head columns ----
  v16h bv[2][2];
#pragma unroll
  for (int nt = 0; nt < 2; ++nt)
#pragma unroll
    for (int kt = 0; kt < 2; ++kt)
      bv[nt][kt] = load_b16(VhT + nt * 16 * NP + kt * 32, NP, lane);
  for (int mt = 0; mt < 4; ++mt) {
    v16h ap0 = load_a16(Ph + mt * 16 * NP, NP, lane);
    v16h ap1 = load_a16(Ph + mt * 16 * NP + 32, NP, lane);
#pragma unroll
    for (int nt = 0; nt < 2; ++nt) {
      v8f o = {};
      o = wmma16(ap0, bv[nt][0], o);
      o = wmma16(ap1, bv[nt][1], o);
#pragma unroll
      for (int r = 0; r < 8; ++r)
        Osh[(mt * 16 + r + hl * 8) * CDIM + h * HD + nt * 16 + ln] = (_Float16)o[r];
    }
  }
  __syncthreads();

  // ---- Output projection: wave h computes output columns [h*32, h*32+32) ----
  float* po = out + (size_t)blockIdx.x * (NTOK * CDIM);
  for (int mt = 0; mt < 4; ++mt) {
    v8f a0 = {}, a1 = {};
#pragma unroll
    for (int kt = 0; kt < 8; ++kt) {
      v16h a  = load_a16(Osh + mt * 16 * CDIM + kt * 32, CDIM, lane);
      v16h b0 = load_b16(pw16 + (size_t)(h * HD) * CDIM + kt * 32, CDIM, lane);
      v16h b1 = load_b16(pw16 + (size_t)(h * HD + 16) * CDIM + kt * 32, CDIM, lane);
      a0 = wmma16(a, b0, a0);
      a1 = wmma16(a, b1, a1);
    }
    const float bb0 = proj_b[h * HD + ln];
    const float bb1 = proj_b[h * HD + 16 + ln];
#pragma unroll
    for (int r = 0; r < 8; ++r) {
      const int row = mt * 16 + r + hl * 8;
      if (row < NTOK) {
        po[row * CDIM + h * HD + ln]      = a0[r] + bb0;
        po[row * CDIM + h * HD + 16 + ln] = a1[r] + bb1;
      }
    }
  }
}

extern "C" void kernel_launch(void* const* d_in, const int* in_sizes, int n_in,
                              void* d_out, int out_size, void* d_ws, size_t ws_size,
                              hipStream_t stream) {
  const float* x1     = (const float*)d_in[0];
  const float* x2     = (const float*)d_in[1];
  const float* qkv_w  = (const float*)d_in[2];
  const float* qkv_b  = (const float*)d_in[3];
  const float* proj_w = (const float*)d_in[4];
  const float* proj_b = (const float*)d_in[5];
  const float* rpb    = (const float*)d_in[6];
  const int*   rel    = (const int*)d_in[7];
  float* outp = (float*)d_out;

  const int B = in_sizes[0] / (NTOK * CDIM);  // 2048

  // Workspace layout: f16 qkv_w (384 KB) | f16 proj_w (128 KB) | bias (128 KB)
  _Float16* w16  = (_Float16*)d_ws;
  _Float16* pw16 = (_Float16*)((char*)d_ws + (size_t)3 * CDIM * CDIM * 2);
  float*    bia  = (float*)((char*)d_ws + (size_t)3 * CDIM * CDIM * 2
                                        + (size_t)CDIM * CDIM * 2);

  winxattn_prep<<<256, 256, 0, stream>>>(qkv_w, proj_w, rpb, rel, w16, pw16, bia);

  const int smem_bytes = 262144;  // 256 KB dynamic LDS (WGP allows 320 KB)
  (void)hipFuncSetAttribute((const void*)winxattn_main,
                            hipFuncAttributeMaxDynamicSharedMemorySize, smem_bytes);
  winxattn_main<<<B, 256, smem_bytes, stream>>>(x1, x2, qkv_b, proj_b,
                                                w16, pw16, bia, outp);
  (void)n_in; (void)out_size; (void)ws_size;
}
